// Mamba2Block_4475355922524
// MI455X (gfx1250) — compile-verified, matrix operand order checked by reading, not run on hardware
//
#include <hip/hip_runtime.h>

// Mamba2 SSD chunk scan for gfx1250 (MI455X), wave32 + v_wmma_f32_16x16x32_bf16.
// Shapes fixed by the reference: B=2, S=2048, H=32, Dh=64, N=128, CHUNK=256, NC=8.
//
// All WMMA operands are stored K-contiguous so each 16x32 bf16 fragment is
// exactly two 16-byte loads per lane (ds_load_b128 / global_load_b128), per the
// CDNA5 fragment layout (K = half*8 + 0..7 and K = 16 + half*8 + 0..7).

#define CS  256
#define NC  8
#define HH  32
#define DH  64
#define NN  128
#define BSZ 2
#define SS  2048

typedef __attribute__((ext_vector_type(16))) __bf16 bf16x16;
typedef __attribute__((ext_vector_type(8)))  __bf16 bf16x8;
typedef __attribute__((ext_vector_type(8)))  float  f32x8;

__device__ __forceinline__ f32x8 wmma_bf16(bf16x16 a, bf16x16 b, f32x8 c) {
  // D = A(16x32) * B(32x16) + C, f32 accumulate
  return __builtin_amdgcn_wmma_f32_16x16x32_bf16(false, a, false, b, (short)0, c, false, false);
}

__device__ __forceinline__ f32x8 zero8() {
  f32x8 z;
#pragma unroll
  for (int i = 0; i < 8; ++i) z[i] = 0.0f;
  return z;
}

// Fragment load from a K-contiguous (row stride rs elements) matrix.
// Works both as A-frag (rows = M) and as B-frag (rows = N) since the CDNA5
// 16-bit A and B layouts use the same per-lane K mapping.
// Two aligned 16-byte vector loads per lane.
__device__ __forceinline__ bf16x16 load_frag(const __bf16* src, int rs, int r0, int c0, int lane) {
  const __bf16* p = src + (size_t)(r0 + (lane & 15)) * rs + c0 + ((lane >> 4) & 1) * 8;
  bf16x8 lo = *(const bf16x8*)p;         // K = half*8 + 0..7
  bf16x8 hi = *(const bf16x8*)(p + 16);  // K = 16 + half*8 + 0..7
  return __builtin_shufflevector(lo, hi, 0, 1, 2, 3, 4, 5, 6, 7,
                                 8, 9, 10, 11, 12, 13, 14, 15);
}

// ---------------------------------------------------------------------------
// Kernel 0: convert B and C (group-shared across heads) to bf16; also emit
// B transposed [n][s] so the states-GEMM A-fragments are K(s)-contiguous.
__global__ __launch_bounds__(256) void k_cvt_bc(const float* __restrict__ Bf,
                                               const float* __restrict__ Cf,
                                               __bf16* __restrict__ Bb,
                                               __bf16* __restrict__ Cb,
                                               __bf16* __restrict__ BT) {
  int i = blockIdx.x * 256 + threadIdx.x;  // grid sized exactly to 2*2048*128
  float bv = Bf[i];
  Bb[i] = (__bf16)bv;
  Cb[i] = (__bf16)Cf[i];
  int b = i >> 18;              // / (2048*128)
  int rem = i & ((1 << 18) - 1);
  int sg = rem >> 7, nr = rem & 127;
  BT[((size_t)(b * NN + nr)) * SS + sg] = (__bf16)bv;
}

// ---------------------------------------------------------------------------
// Kernel 1: dt softplus + clip, per-chunk cumsum of A*dt (LDS scan),
// x = h*dt -> bf16 stored TRANSPOSED per tile: xbf[tile][d][s].
__global__ __launch_bounds__(256) void k_prep(const float* __restrict__ hs,
                                              const float* __restrict__ dt,
                                              const float* __restrict__ A,
                                              const float* __restrict__ dtb,
                                              float* __restrict__ cumA,
                                              __bf16* __restrict__ xbf) {
  __shared__ float sbuf[CS];
  int tid = threadIdx.x;
  int h = blockIdx.x & 31, n = (blockIdx.x >> 5) & 7, b = blockIdx.x >> 8;
  int sg = n * CS + tid;
  float z = dt[(size_t)(b * SS + sg) * HH + h] + dtb[h];
  float dtv = (z > 20.f) ? z : log1pf(expf(z));   // stable softplus
  dtv = fmaxf(dtv, 0.f);                          // clip
  sbuf[tid] = A[h] * dtv;
  __syncthreads();
#pragma unroll
  for (int off = 1; off < CS; off <<= 1) {        // Hillis-Steele inclusive scan
    float v = (tid >= off) ? sbuf[tid - off] : 0.f;
    __syncthreads();
    sbuf[tid] += v;
    __syncthreads();
  }
  size_t tile = (size_t)(b * NC + n) * HH + h;
  cumA[tile * CS + tid] = sbuf[tid];
  const float* hrow = hs + ((size_t)(b * SS + sg) * HH + h) * DH;
  __bf16* xt = xbf + tile * CS * DH;              // [d][s]
#pragma unroll 8
  for (int d = 0; d < DH; ++d) xt[d * CS + tid] = (__bf16)(hrow[d] * dtv);
}

// ---------------------------------------------------------------------------
// Kernel 2: per-chunk states  S[nr][d] = sum_s B[s,nr] * (dec[s]*x[s,d])
// Decay folded into an LDS decayed-x buffer -> both fragments are pure b128.
// One block per (b, chunk, head); 8 waves, each owns a 16-row slab of N=128.
__global__ __launch_bounds__(256) void k_states(const __bf16* __restrict__ xbf,
                                                const __bf16* __restrict__ BT,
                                                const float* __restrict__ cumA,
                                                float* __restrict__ states) {
  __shared__ __align__(16) __bf16 lXd[DH * CS];  // 32 KB: dec[s]*x, layout [d][s]
  __shared__ float lcumA[CS];
  __shared__ float ldec[CS];
  int tid = threadIdx.x, lane = tid & 31, w = tid >> 5;
  int h = blockIdx.x & 31, n = (blockIdx.x >> 5) & 7, b = blockIdx.x >> 8;
  size_t tile = (size_t)(b * NC + n) * HH + h;

  lcumA[tid] = cumA[tile * CS + tid];
  __syncthreads();
  ldec[tid] = expf(lcumA[CS - 1] - lcumA[tid]);
  __syncthreads();
  const __bf16* xt = xbf + tile * CS * DH;  // [d][s]
#pragma unroll
  for (int i = tid; i < DH * CS; i += 256) {
    int s = i & (CS - 1);
    lXd[i] = (__bf16)((float)xt[i] * ldec[s]);
  }
  __syncthreads();

  const __bf16* BTb = BT + (size_t)b * NN * SS;  // [nr][sg]
  int nr0 = w * 16, sbase = n * CS;
  f32x8 acc[4];
#pragma unroll
  for (int i = 0; i < 4; ++i) acc[i] = zero8();

  for (int s0 = 0; s0 < CS; s0 += 32) {
    __builtin_prefetch(BTb + (size_t)nr0 * SS + sbase + s0, 0, 1);
    // A-frag: B^T[nr][s]  (K = s contiguous)
    bf16x16 af = load_frag(BTb, SS, nr0, sbase + s0, lane);
#pragma unroll
    for (int d4 = 0; d4 < 4; ++d4)
      acc[d4] = wmma_bf16(af, load_frag(lXd, CS, d4 * 16, s0, lane), acc[d4]);
  }
  // D-layout writeback: lane = N col d, VGPR r = row (nr0 + r + 8*half)
  float* so = states + tile * NN * DH;
  int half = (lane >> 4) & 1, ncol = lane & 15;
#pragma unroll
  for (int d4 = 0; d4 < 4; ++d4)
#pragma unroll
    for (int r = 0; r < 8; ++r)
      so[(size_t)(nr0 + r + 8 * half) * DH + d4 * 16 + ncol] = acc[d4][r];
}

// ---------------------------------------------------------------------------
// Kernel 3: sequential 8-chunk scan per (b,h). Emits the EXCLUSIVE prefix
// state (state entering chunk n) as bf16 TRANSPOSED [d][nr] for k_y's b128
// fragment loads, and the final state (f32) to d_out tail.
__global__ __launch_bounds__(256) void k_scan(const float* __restrict__ cumA,
                                              const float* __restrict__ states,
                                              __bf16* __restrict__ stbf,
                                              float* __restrict__ fs) {
  int tid = threadIdx.x;
  int h = blockIdx.x & 31, b = blockIdx.x >> 5;
  float run[32];
#pragma unroll
  for (int j = 0; j < 32; ++j) run[j] = 0.f;
  for (int n = 0; n < NC; ++n) {
    size_t tile = (size_t)(b * NC + n) * HH + h;
    float dec = expf(cumA[tile * CS + (CS - 1)]);
    const float* st = states + tile * NN * DH;   // [nr][d]
    __bf16* sb = stbf + tile * NN * DH;          // [d][nr]
#pragma unroll
    for (int j = 0; j < 32; ++j) {
      int e = tid + 256 * j;
      int nr = e >> 6, d = e & 63;
      sb[d * NN + nr] = (__bf16)run[j];          // exclusive prefix for chunk n
      run[j] = run[j] * dec + st[e];
    }
  }
  // final_state layout: (b, H, Dh, N); our element e = nr*DH + d
#pragma unroll
  for (int j = 0; j < 32; ++j) {
    int e = tid + 256 * j;
    int nr = e >> 6, d = e & 63;
    fs[((size_t)(b * HH + h) * DH + d) * NN + nr] = run[j];
  }
}

// ---------------------------------------------------------------------------
// Kernel 4: Y = sdo * (C @ S_prev) + (L*(C@B^T)) @ x + D*hidden
// One block per (b, chunk, head); 8 waves, each owns a 32-row l-slab.
__global__ __launch_bounds__(256) void k_y(const __bf16* __restrict__ xbf,
                                           const __bf16* __restrict__ Bbf,
                                           const __bf16* __restrict__ Cbf,
                                           const float* __restrict__ cumA,
                                           const __bf16* __restrict__ stbf,
                                           const float* __restrict__ hs,
                                           const float* __restrict__ D,
                                           float* __restrict__ Y) {
  __shared__ __align__(16) __bf16 lXT[DH * CS];     // 32 KB, x transposed [d][s]
  __shared__ __align__(16) __bf16 lG[8 * 32 * 32];  // 16 KB per-wave M-tile staging
  __shared__ float lcumA[CS];
  __shared__ float lsdo[CS];
  int tid = threadIdx.x, lane = tid & 31, w = tid >> 5;
  int h = blockIdx.x & 31, n = (blockIdx.x >> 5) & 7, b = blockIdx.x >> 8;
  size_t tile = (size_t)(b * NC + n) * HH + h;

  lcumA[tid] = cumA[tile * CS + tid];
  const unsigned int* xs = (const unsigned int*)(xbf + tile * CS * DH);
  unsigned int* xd = (unsigned int*)lXT;
#pragma unroll
  for (int i = tid; i < CS * DH / 2; i += 256) xd[i] = xs[i];
  __syncthreads();
  lsdo[tid] = expf(lcumA[tid]);
  __syncthreads();

  const __bf16* Cc = Cbf + (size_t)(b * SS + n * CS) * NN;   // [l][nn]
  const __bf16* Bc = Bbf + (size_t)(b * SS + n * CS) * NN;   // [k][nn]
  const __bf16* stp = stbf + tile * NN * DH;                 // [d][nr]
  int lb = w * 32;
  int half = (lane >> 4) & 1, ncol = lane & 15;

  __builtin_prefetch(Cc + (size_t)lb * NN, 0, 1);
  __builtin_prefetch(stp, 0, 1);

  // C slab as WMMA-A fragments: rows l, K over N=128 (4 chunks of 32)
  bf16x16 cfrag[2][4];
#pragma unroll
  for (int lt = 0; lt < 2; ++lt)
#pragma unroll
    for (int kc = 0; kc < 4; ++kc)
      cfrag[lt][kc] = load_frag(Cc, NN, lb + lt * 16, kc * 32, lane);

  f32x8 acc[2][4];
#pragma unroll
  for (int lt = 0; lt < 2; ++lt)
#pragma unroll
    for (int d4 = 0; d4 < 4; ++d4) acc[lt][d4] = zero8();

  // ---- Y_off = C @ S_prev, then scale rows by exp(cumA[l]) ----
#pragma unroll
  for (int kc = 0; kc < 4; ++kc)
#pragma unroll
    for (int d4 = 0; d4 < 4; ++d4) {
      bf16x16 sf = load_frag(stp, NN, d4 * 16, kc * 32, lane);  // B[K=nr][N=d]
      acc[0][d4] = wmma_bf16(cfrag[0][kc], sf, acc[0][d4]);
      acc[1][d4] = wmma_bf16(cfrag[1][kc], sf, acc[1][d4]);
    }
#pragma unroll
  for (int lt = 0; lt < 2; ++lt)
#pragma unroll
    for (int r = 0; r < 8; ++r) {
      float sc = lsdo[lb + lt * 16 + r + 8 * half];
#pragma unroll
      for (int d4 = 0; d4 < 4; ++d4) acc[lt][d4][r] *= sc;
    }

  // ---- Y_diag: causal tiles only (wave w needs k-tiles 0..w) ----
  __bf16* gb = lG + w * 32 * 32;
  for (int kt = 0; kt <= w; ++kt) {
    int k0 = kt * 32;
    f32x8 g[2][2];
#pragma unroll
    for (int lt = 0; lt < 2; ++lt) { g[lt][0] = zero8(); g[lt][1] = zero8(); }
    // G = C @ B^T : B-frag (K=nn, N=kk) = B[kk][nn], K contiguous in memory
#pragma unroll
    for (int kc = 0; kc < 4; ++kc) {
      bf16x16 b0 = load_frag(Bc, NN, k0,      kc * 32, lane);
      bf16x16 b1 = load_frag(Bc, NN, k0 + 16, kc * 32, lane);
#pragma unroll
      for (int lt = 0; lt < 2; ++lt) {
        g[lt][0] = wmma_bf16(cfrag[lt][kc], b0, g[lt][0]);
        g[lt][1] = wmma_bf16(cfrag[lt][kc], b1, g[lt][1]);
      }
    }
    // M = G * L (causal decay mask), staged through per-wave LDS to re-layout D->A
#pragma unroll
    for (int lt = 0; lt < 2; ++lt)
#pragma unroll
      for (int ks = 0; ks < 2; ++ks)
#pragma unroll
        for (int r = 0; r < 8; ++r) {
          int l  = lb + lt * 16 + r + 8 * half;
          int kk = k0 + ks * 16 + ncol;
          float v = g[lt][ks][r];
          v = (kk <= l) ? v * expf(lcumA[l] - lcumA[kk]) : 0.f;
          gb[(lt * 16 + r + 8 * half) * 32 + ks * 16 + ncol] = (__bf16)v;
        }
    __builtin_amdgcn_wave_barrier();  // same-wave LDS RAW; DS ops are in-order
#pragma unroll
    for (int lt = 0; lt < 2; ++lt) {
      bf16x16 mf = load_frag(gb, 32, lt * 16, 0, lane);
#pragma unroll
      for (int d4 = 0; d4 < 4; ++d4)
        acc[lt][d4] = wmma_bf16(mf, load_frag(lXT, CS, d4 * 16, k0, lane), acc[lt][d4]);
    }
  }

  // ---- epilogue: + D * hidden, store Y[b, s, h, d] ----
  float Dv = D[h];
#pragma unroll
  for (int lt = 0; lt < 2; ++lt)
#pragma unroll
    for (int d4 = 0; d4 < 4; ++d4)
#pragma unroll
      for (int r = 0; r < 8; ++r) {
        int l = lb + lt * 16 + r + 8 * half;
        int d = d4 * 16 + ncol;
        size_t gi = ((size_t)(b * SS + n * CS + l) * HH + h) * DH + d;
        Y[gi] = acc[lt][d4][r] + Dv * hs[gi];
      }
}

// ---------------------------------------------------------------------------
extern "C" void kernel_launch(void* const* d_in, const int* in_sizes, int n_in,
                              void* d_out, int out_size, void* d_ws, size_t ws_size,
                              hipStream_t stream) {
  (void)in_sizes; (void)n_in; (void)out_size; (void)ws_size;
  const float* hidden = (const float*)d_in[0];   // (2,2048,32,64)
  const float* dt     = (const float*)d_in[1];   // (2,2048,32)
  const float* A      = (const float*)d_in[2];   // (32)
  const float* Bm     = (const float*)d_in[3];   // (2,2048,1,128)
  const float* Cm     = (const float*)d_in[4];   // (2,2048,1,128)
  const float* Dp     = (const float*)d_in[5];   // (32)
  const float* dtb    = (const float*)d_in[6];   // (32)

  float* Y  = (float*)d_out;                                // 2*2048*32*64
  float* FS = (float*)d_out + (size_t)BSZ * SS * HH * DH;   // 2*32*64*128

  // Workspace layout (bytes)
  char* ws = (char*)d_ws;
  float*  cumA   = (float*)ws;                               // 512 KB
  __bf16* xbf    = (__bf16*)(ws + (512u << 10));             // 16 MB (x^T per tile)
  __bf16* Bbf    = (__bf16*)(ws + (512u << 10) + (16u << 20));          // 1 MB
  __bf16* Cbf    = (__bf16*)(ws + (512u << 10) + (17u << 20));          // 1 MB
  __bf16* BT     = (__bf16*)(ws + (512u << 10) + (18u << 20));          // 1 MB
  float*  states = (float*)(ws + (512u << 10) + (19u << 20));           // 16 MB
  __bf16* stbf   = (__bf16*)(ws + (512u << 10) + (35u << 20));          // 8 MB

  k_cvt_bc<<<dim3((BSZ * SS * NN) / 256), dim3(256), 0, stream>>>(Bm, Cm, Bbf, Cbf, BT);
  k_prep  <<<dim3(BSZ * NC * HH), dim3(256), 0, stream>>>(hidden, dt, A, dtb, cumA, xbf);
  k_states<<<dim3(BSZ * NC * HH), dim3(256), 0, stream>>>(xbf, BT, cumA, states);
  k_scan  <<<dim3(BSZ * HH),      dim3(256), 0, stream>>>(cumA, states, stbf, FS);
  k_y     <<<dim3(BSZ * NC * HH), dim3(256), 0, stream>>>(xbf, Bbf, Cbf, cumA, stbf,
                                                          hidden, Dp, Y);
}
